// SSM_15461882266199
// MI455X (gfx1250) — matrix operand order repaired and checked by reading, not compiled, not run on hardware
//
#include <hip/hip_runtime.h>
#include <hip/hip_bf16.h>
#include <math.h>

// ---------------- types ----------------
typedef __bf16 bf16;
typedef __attribute__((ext_vector_type(16))) __bf16 v16bf;
typedef __attribute__((ext_vector_type(8)))  __bf16 v8bf;
typedef __attribute__((ext_vector_type(4)))  __bf16 v4bf;
typedef __attribute__((ext_vector_type(8)))  float  v8f;

union VF8  { v8f v;  float f[8]; };
union VB16 { v16bf v; v8bf h[2]; };
union PK8  { v8bf v; bf16 e[8]; };
union PK4  { v4bf v; bf16 e[4]; };

// ---------------- model dims ----------------
#define DIMC 2048
#define HIDC 4096
#define HC   64
#define PC   64
#define NC   128
#define KC   4
#define QC   128
#define CONVD 4352          // HID + 2*N
#define DINC  8512          // 2*HID + 2*N + H
#define LTOT  4096
#define BSZ   2
#define TOKS  (BSZ*LTOT)    // 8192
#define NCH   (LTOT/QC)     // 32 chunks per batch

// ---------------- WMMA helpers ----------------
__device__ __forceinline__ v8f wmma_bf16(v16bf a, v16bf b, v8f c) {
  return __builtin_amdgcn_wmma_f32_16x16x32_bf16(false, a, false, b, (short)0, c,
                                                 false, false);
}

__device__ __forceinline__ v8f vzero8() {
  v8f v = {0.f,0.f,0.f,0.f,0.f,0.f,0.f,0.f};
  return v;
}

// A fragment: 16x32 tile, row-major bf16, row stride ld.
// lanes 0-15: K 0..7 & 16..23; lanes 16-31: K 8..15 & 24..31 (ISA 7.12.2).
__device__ __forceinline__ v16bf load_fragA(const bf16* base, int ld, int lane) {
  int half = lane >> 4;
  int m    = lane & 15;
  const bf16* p = base + (size_t)m * ld;
  VB16 u;
  u.h[0] = *(const v8bf*)(p + half * 8);
  u.h[1] = *(const v8bf*)(p + 16 + half * 8);
  return u.v;
}

// B operand: 32x16 (KxN) stored as [N][K] row-major; lanes 0-15 K 0..7 & 16..23,
// lanes 16-31 K 8..15 & 24..31 (same as A), N = lane&15.
__device__ __forceinline__ v16bf load_fragB(const bf16* baseT, int ld, int lane) {
  int half = lane >> 4;
  int n    = lane & 15;
  const bf16* p = baseT + (size_t)n * ld;
  VB16 u;
  u.h[0] = *(const v8bf*)(p + half * 8);
  u.h[1] = *(const v8bf*)(p + 16 + half * 8);
  return u.v;
}

__device__ __forceinline__ void pack_store8(bf16* dst, const float4& lo, const float4& hi) {
  PK8 pk;
  pk.e[0] = (bf16)lo.x; pk.e[1] = (bf16)lo.y; pk.e[2] = (bf16)lo.z; pk.e[3] = (bf16)lo.w;
  pk.e[4] = (bf16)hi.x; pk.e[5] = (bf16)hi.y; pk.e[6] = (bf16)hi.z; pk.e[7] = (bf16)hi.w;
  *(v8bf*)dst = pk.v;
}

// ---------------- Kernel 1/6: double-buffered bf16 WMMA GEMM ----------------
// C[M][Nn] = A[M][Kd] * B[Nn][Kd]^T (row-major f32 in, f32 out, bf16 math).
// M multiple of 256, Kd multiple of 32; Nn guarded by clamp+store-mask.
#define BM 256
#define BN 128
#define BK 32

__global__ __launch_bounds__(256)
void gemm_bf16_kernel(const float* __restrict__ A, const float* __restrict__ B,
                      float* __restrict__ C, int M, int Nn, int Kd) {
  __shared__ bf16 As[2][BM][BK];   // 2 x 16KB
  __shared__ bf16 Bs[2][BN][BK];   // 2 x  8KB

  const int tid  = threadIdx.x;
  const int lane = tid & 31;
  const int wave = tid >> 5;
  const int m0 = blockIdx.y * BM;
  const int n0 = blockIdx.x * BN;
  const int wm = (wave >> 1) * 64;   // 4 wave rows x 64
  const int wn = (wave & 1) * 64;    // 2 wave cols x 64

  VF8 acc[4][4];
  #pragma unroll
  for (int a = 0; a < 4; ++a)
    #pragma unroll
    for (int b = 0; b < 4; ++b) acc[a][b].v = vzero8();

  // staging registers: A 4 chunks of 8, B 2 chunks of 8
  float4 ra[4][2], rb[2][2];

  // per-thread chunk coordinates (8 consecutive K elements per chunk)
  int arow[4], acol[4], brow[2], bcol[2];
  #pragma unroll
  for (int k = 0; k < 4; ++k) {
    int ci = tid + k * 256;        // 0..1023  (BM*BK/8 = 1024 chunks)
    arow[k] = ci >> 2;             // 4 chunks per 32-wide row
    acol[k] = (ci & 3) * 8;
  }
  #pragma unroll
  for (int k = 0; k < 2; ++k) {
    int ci = tid + k * 256;        // 0..511   (BN*BK/8 = 512 chunks)
    brow[k] = ci >> 2;
    bcol[k] = (ci & 3) * 8;
  }

  auto load_tile = [&](int k0) {
    #pragma unroll
    for (int k = 0; k < 4; ++k) {
      const float* p = A + (size_t)(m0 + arow[k]) * Kd + k0 + acol[k];
      ra[k][0] = *(const float4*)(p);
      ra[k][1] = *(const float4*)(p + 4);
    }
    #pragma unroll
    for (int k = 0; k < 2; ++k) {
      int gn = n0 + brow[k];
      if (gn >= Nn) gn = Nn - 1;   // clamp: branchless, junk cols never stored
      const float* p = B + (size_t)gn * Kd + k0 + bcol[k];
      rb[k][0] = *(const float4*)(p);
      rb[k][1] = *(const float4*)(p + 4);
    }
  };
  auto store_tile = [&](int buf) {
    #pragma unroll
    for (int k = 0; k < 4; ++k)
      pack_store8(&As[buf][arow[k]][acol[k]], ra[k][0], ra[k][1]);
    #pragma unroll
    for (int k = 0; k < 2; ++k)
      pack_store8(&Bs[buf][brow[k]][bcol[k]], rb[k][0], rb[k][1]);
  };

  const int nsteps = Kd / BK;
  load_tile(0);
  store_tile(0);
  __syncthreads();

  for (int s = 0; s < nsteps; ++s) {
    const int cur = s & 1;
    if (s + 1 < nsteps) load_tile((s + 1) * BK);   // in flight during WMMAs

    v16bf af[4];
    #pragma unroll
    for (int a = 0; a < 4; ++a)
      af[a] = load_fragA(&As[cur][wm + a * 16][0], BK, lane);
    #pragma unroll
    for (int b = 0; b < 4; ++b) {
      v16bf bv = load_fragB(&Bs[cur][wn + b * 16][0], BK, lane);
      #pragma unroll
      for (int a = 0; a < 4; ++a)
        acc[a][b].v = wmma_bf16(af[a], bv, acc[a][b].v);
    }
    __syncthreads();
    if (s + 1 < nsteps) {
      store_tile(cur ^ 1);
      __syncthreads();
    }
  }

  // store: per v8f reg r -> (m = r + 8*half, n = lane&15)
  const int half = lane >> 4, nl = lane & 15;
  #pragma unroll
  for (int a = 0; a < 4; ++a)
    #pragma unroll
    for (int b = 0; b < 4; ++b) {
      int gn = n0 + wn + b * 16 + nl;
      if (gn >= Nn) continue;
      #pragma unroll
      for (int r = 0; r < 8; ++r) {
        int gm = m0 + wm + a * 16 + half * 8 + r;
        C[(size_t)gm * Nn + gn] = acc[a][b].f[r];
      }
    }
}

// ---------------- Kernel 2: depthwise causal conv (K=4) + SiLU ----------------
__global__ __launch_bounds__(256)
void conv_silu_kernel(const float* __restrict__ zx, const float* __restrict__ w,
                      float* __restrict__ out) {
  size_t idx = (size_t)blockIdx.x * 256 + threadIdx.x;
  if (idx >= (size_t)TOKS * CONVD) return;
  int c = (int)(idx % CONVD);
  size_t t = idx / CONVD;
  int b = (int)(t >> 12);
  int l = (int)(t & 4095);
  float acc = 0.f;
  #pragma unroll
  for (int j = 0; j < KC; ++j) {
    int ls = l - (KC - 1) + j;
    if (ls >= 0)
      acc += zx[((size_t)b * LTOT + ls) * DINC + HIDC + c] * w[c * KC + j];
  }
  out[idx] = acc / (1.f + __expf(-acc));
}

// ---------------- Kernel 3: dt = softplus(dt_raw+bias), cum = cumsum(dt*A) ----
__global__ __launch_bounds__(128)
void dt_cum_kernel(const float* __restrict__ zx, const float* __restrict__ dt_bias,
                   const float* __restrict__ A_log,
                   float* __restrict__ dt_out, float* __restrict__ cum_out) {
  int idx = blockIdx.x * 128 + threadIdx.x;   // (b, chunk, h): 2*32*64 = 4096
  if (idx >= BSZ * NCH * HC) return;
  int h = idx & 63;
  int c = (idx >> 6) & (NCH - 1);
  int b = idx >> 11;
  const float A    = -__expf(A_log[h]);
  const float bias = dt_bias[h];
  float run = 0.f;
  for (int q = 0; q < QC; ++q) {
    size_t t = (size_t)b * LTOT + c * QC + q;
    float raw = zx[t * DINC + HIDC + CONVD + h] + bias;
    float dt = (raw > 20.f) ? raw : log1pf(__expf(raw));
    run += dt * A;
    dt_out[t * HC + h]  = dt;
    cum_out[t * HC + h] = run;
  }
}

// ---------------- Kernel 4: chunked SSD scan (one block per (b,h)) ----------
__global__ __launch_bounds__(256)
void ssd_kernel(const float* __restrict__ xbc,   // [TOKS][CONVD]
                const float* __restrict__ dt,    // [TOKS][H]
                const float* __restrict__ cum,   // [TOKS][H]
                const float* __restrict__ Dv,    // [H]
                float* __restrict__ y)           // [TOKS][HID]
{
  extern __shared__ __align__(16) char smem[];
  bf16* Cs   = (bf16*)smem;                 // 128*128
  bf16* Bs   = Cs  + 128 * 128;
  bf16* BTs  = Bs  + 128 * 128;
  bf16* Ms   = BTs + 128 * 128;
  bf16* xsT  = Ms  + 128 * 128;             // 64*128
  bf16* Sbf  = xsT + 64 * 128;              // 64*128
  float* Sst = (float*)(Sbf + 64 * 128);    // 64*128 f32
  float* cumS = Sst + 64 * 128;
  float* dtS  = cumS + 128;

  const int tid  = threadIdx.x;
  const int lane = tid & 31;
  const int wave = tid >> 5;
  const int half = lane >> 4;
  const int nl   = lane & 15;
  const int b = blockIdx.x >> 6;
  const int h = blockIdx.x & 63;
  const float Dh = Dv[h];

  for (int i = tid; i < 64 * 128; i += 256) { Sst[i] = 0.f; Sbf[i] = (bf16)0.f; }
  __syncthreads();

  for (int c = 0; c < NCH; ++c) {
    const size_t tok0 = (size_t)b * LTOT + (size_t)c * QC;

    // ---- load chunk: B/C rows vectorized, xs transposed, cum/dt
    #pragma unroll 2
    for (int i = tid; i < (128 * 128) / 4; i += 256) {   // 16 iters
      int q  = i >> 5;
      int n4 = (i & 31) * 4;
      const float* row = xbc + (tok0 + q) * CONVD;
      float4 fb = *(const float4*)(row + HIDC + n4);
      float4 fc = *(const float4*)(row + HIDC + NC + n4);
      PK4 pb, pc;
      pb.e[0] = (bf16)fb.x; pb.e[1] = (bf16)fb.y; pb.e[2] = (bf16)fb.z; pb.e[3] = (bf16)fb.w;
      pc.e[0] = (bf16)fc.x; pc.e[1] = (bf16)fc.y; pc.e[2] = (bf16)fc.z; pc.e[3] = (bf16)fc.w;
      *(v4bf*)(Bs + q * 128 + n4) = pb.v;
      *(v4bf*)(Cs + q * 128 + n4) = pc.v;
    }
    #pragma unroll 2
    for (int i = tid; i < (128 * 64) / 4; i += 256) {    // 8 iters
      int q  = i >> 4;
      int p4 = (i & 15) * 4;
      float4 f = *(const float4*)(xbc + (tok0 + q) * CONVD + h * PC + p4);
      xsT[(p4 + 0) * 128 + q] = (bf16)f.x;
      xsT[(p4 + 1) * 128 + q] = (bf16)f.y;
      xsT[(p4 + 2) * 128 + q] = (bf16)f.z;
      xsT[(p4 + 3) * 128 + q] = (bf16)f.w;
    }
    for (int i = tid; i < 128; i += 256) {
      cumS[i] = cum[(tok0 + i) * HC + h];
      dtS[i]  = dt[(tok0 + i) * HC + h];
    }
    __syncthreads();
    const float cumLast = cumS[127];

    // ---- BTs[n][q] = B[q][n] * exp(cumLast - cum_q) * dt_q
    for (int i = tid; i < 128 * 128; i += 256) {
      int n = i >> 7, q = i & 127;
      float wdec = __expf(cumLast - cumS[q]) * dtS[q];
      BTs[n * 128 + q] = (bf16)((float)Bs[q * 128 + n] * wdec);
    }
    __syncthreads();

    // ---- CB[i][j] = sum_n C[i,n]B[j,n]; M = CB*exp(cum_i-cum_j)*dt_j (i>=j)
    {
      const int ti = wave;
      VF8 acc[8];
      #pragma unroll
      for (int tj = 0; tj < 8; ++tj) acc[tj].v = vzero8();
      #pragma unroll
      for (int k0 = 0; k0 < 128; k0 += 32) {
        v16bf a = load_fragA(Cs + (size_t)(ti * 16) * 128 + k0, 128, lane);
        for (int tj = 0; tj <= ti; ++tj) {
          v16bf bb = load_fragB(Bs + (size_t)(tj * 16) * 128 + k0, 128, lane);
          acc[tj].v = wmma_bf16(a, bb, acc[tj].v);
        }
      }
      #pragma unroll
      for (int tj = 0; tj < 8; ++tj)
        #pragma unroll
        for (int r = 0; r < 8; ++r) {
          int i = ti * 16 + half * 8 + r;
          int j = tj * 16 + nl;
          float v = 0.f;
          if (tj <= ti && i >= j)
            v = acc[tj].f[r] * __expf(cumS[i] - cumS[j]) * dtS[j];
          Ms[i * 128 + j] = (bf16)v;
        }
    }
    __syncthreads();

    // ---- Yd = M @ xs ; Yoff = exp(cum_i)*(C @ Sprev^T) ; y = Yd+Yoff+D*xs
    {
      const int ti = wave;
      VF8 accd[4], acco[4];
      #pragma unroll
      for (int tp = 0; tp < 4; ++tp) { accd[tp].v = vzero8(); acco[tp].v = vzero8(); }
      #pragma unroll
      for (int k0 = 0; k0 < 128; k0 += 32) {
        v16bf a = load_fragA(Ms + (size_t)(ti * 16) * 128 + k0, 128, lane);
        #pragma unroll
        for (int tp = 0; tp < 4; ++tp) {
          v16bf bb = load_fragB(xsT + (size_t)(tp * 16) * 128 + k0, 128, lane);
          accd[tp].v = wmma_bf16(a, bb, accd[tp].v);
        }
      }
      #pragma unroll
      for (int k0 = 0; k0 < 128; k0 += 32) {
        v16bf a = load_fragA(Cs + (size_t)(ti * 16) * 128 + k0, 128, lane);
        #pragma unroll
        for (int tp = 0; tp < 4; ++tp) {
          v16bf bb = load_fragB(Sbf + (size_t)(tp * 16) * 128 + k0, 128, lane);
          acco[tp].v = wmma_bf16(a, bb, acco[tp].v);
        }
      }
      #pragma unroll
      for (int tp = 0; tp < 4; ++tp)
        #pragma unroll
        for (int r = 0; r < 8; ++r) {
          int i = ti * 16 + half * 8 + r;
          int p = tp * 16 + nl;
          float val = accd[tp].f[r] + __expf(cumS[i]) * acco[tp].f[r]
                      + Dh * (float)xsT[p * 128 + i];
          y[(tok0 + i) * HIDC + h * PC + p] = val;
        }
    }
    __syncthreads();   // all Sbf reads done before update

    // ---- states[p][n] = sum_q xs[q,p]*(wdec_q B[q,n]); S = S*decay + states
    {
      const int pw = wave >> 1;
      const int nb = (wave & 1) * 4;
      VF8 acc[4];
      #pragma unroll
      for (int t = 0; t < 4; ++t) acc[t].v = vzero8();
      #pragma unroll
      for (int k0 = 0; k0 < 128; k0 += 32) {
        v16bf a = load_fragA(xsT + (size_t)(pw * 16) * 128 + k0, 128, lane);
        #pragma unroll
        for (int t = 0; t < 4; ++t) {
          v16bf bb = load_fragB(BTs + (size_t)((nb + t) * 16) * 128 + k0, 128, lane);
          acc[t].v = wmma_bf16(a, bb, acc[t].v);
        }
      }
      const float cdec = __expf(cumLast);
      #pragma unroll
      for (int t = 0; t < 4; ++t)
        #pragma unroll
        for (int r = 0; r < 8; ++r) {
          int p = pw * 16 + half * 8 + r;
          int n = (nb + t) * 16 + nl;
          float s = Sst[p * 128 + n] * cdec + acc[t].f[r];
          Sst[p * 128 + n] = s;
          Sbf[p * 128 + n] = (bf16)s;
        }
    }
    __syncthreads();
  }
}

// ---------------- Kernel 5: yz = y*silu(z); RMS-norm; * norm_w ----------------
__global__ __launch_bounds__(256)
void rms_kernel(const float* __restrict__ y, const float* __restrict__ zx,
                const float* __restrict__ norm_w, float* __restrict__ yn) {
  const size_t t = blockIdx.x;
  const int tid = threadIdx.x, lane = tid & 31, wave = tid >> 5;
  __shared__ float red[8];
  float vals[16];
  float ssum = 0.f;
  #pragma unroll
  for (int i = 0; i < 16; ++i) {
    int c = tid + i * 256;
    float yy = y[t * HIDC + c];
    float z  = zx[t * DINC + c];
    float v  = yy * (z / (1.f + __expf(-z)));
    vals[i] = v;
    ssum += v * v;
  }
  #pragma unroll
  for (int off = 16; off > 0; off >>= 1) ssum += __shfl_xor(ssum, off, 32);
  if (lane == 0) red[wave] = ssum;
  __syncthreads();
  float total = 0.f;
  #pragma unroll
  for (int w = 0; w < 8; ++w) total += red[w];
  const float scale = rsqrtf(total / (float)HIDC + 1e-5f);
  #pragma unroll
  for (int i = 0; i < 16; ++i) {
    int c = tid + i * 256;
    yn[t * HIDC + c] = vals[i] * scale * norm_w[c];
  }
}

// ---------------- launch ----------------
extern "C" void kernel_launch(void* const* d_in, const int* in_sizes, int n_in,
                              void* d_out, int out_size, void* d_ws, size_t ws_size,
                              hipStream_t stream) {
  const float* x       = (const float*)d_in[0];
  const float* W_in    = (const float*)d_in[1];
  const float* conv_w  = (const float*)d_in[2];
  const float* dt_bias = (const float*)d_in[3];
  const float* A_log   = (const float*)d_in[4];
  const float* Dv      = (const float*)d_in[5];
  const float* norm_w  = (const float*)d_in[6];
  const float* W_out   = (const float*)d_in[7];
  float* out = (float*)d_out;

  float* zx   = (float*)d_ws;                        // [8192][8512]
  float* xbc  = zx   + (size_t)TOKS * DINC;          // [8192][4352]
  float* dtb  = xbc  + (size_t)TOKS * CONVD;         // [8192][64]
  float* cumb = dtb  + (size_t)TOKS * HC;            // [8192][64]
  float* yb   = cumb + (size_t)TOKS * HC;            // [8192][4096]
  float* ynb  = xbc;  // reuse conv scratch after ssd_kernel consumed it

  // 1) zxbcdt = x @ W_in^T   (M=8192, N=8512, K=2048)
  gemm_bf16_kernel<<<dim3((DINC + BN - 1) / BN, TOKS / BM), 256, 0, stream>>>(
      x, W_in, zx, TOKS, DINC, DIMC);

  // 2) causal depthwise conv + SiLU
  {
    size_t tot = (size_t)TOKS * CONVD;
    conv_silu_kernel<<<(unsigned)((tot + 255) / 256), 256, 0, stream>>>(
        zx, conv_w, xbc);
  }

  // 3) dt / cumulative decay
  dt_cum_kernel<<<(BSZ * NCH * HC + 127) / 128, 128, 0, stream>>>(
      zx, dt_bias, A_log, dtb, cumb);

  // 4) chunked SSD scan
  {
    size_t smemB = (size_t)(4 * 128 * 128 + 2 * 64 * 128) * sizeof(bf16)
                 + (size_t)(64 * 128 + 256) * sizeof(float);   // ~193 KB
    ssd_kernel<<<BSZ * HC, 256, smemB, stream>>>(xbc, dtb, cumb, Dv, yb);
  }

  // 5) gate + RMS norm
  rms_kernel<<<TOKS, 256, 0, stream>>>(yb, zx, norm_w, ynb);

  // 6) out = yn @ W_out^T   (M=8192, N=2048, K=4096)
  gemm_bf16_kernel<<<dim3(DIMC / BN, TOKS / BM), 256, 0, stream>>>(
      ynb, W_out, out, TOKS, DIMC, HIDC);
}
